// DepthAttnLayer_87771951661504
// MI455X (gfx1250) — matrix operand order repaired and checked by reading, not compiled, not run on hardware
//
#include <hip/hip_runtime.h>
#include <math.h>

#define EMBED 256
#define HEADS 8
#define HDIM  32
#define TGT   32400
#define KLEN  16896

typedef unsigned short ushort_t;
typedef unsigned int   uint_t;

typedef __attribute__((ext_vector_type(16))) __bf16 v16bf;
typedef __attribute__((ext_vector_type(8)))  float  v8f;

union FragBF { v16bf v; uint_t u[8]; };

static __device__ __forceinline__ ushort_t bf_from_bits(uint_t u) {
    // fp32 bits -> bf16 with round-to-nearest-even
    return (ushort_t)((u + 0x7FFFu + ((u >> 16) & 1u)) >> 16);
}
static __device__ __forceinline__ ushort_t f2bf(float f) {
    return bf_from_bits(__float_as_uint(f));
}

// Registers holding one staged K-tile (A: 128x32, B(W): 32x64)
struct StageRegs {
    uint4  a[4];   // fp32 path: 4 raw floats each; bf16 path: .x/.y = 4 bf16
    float4 b[2];
};

// ---------------------------------------------------------------------------
// Stage phase 1: global -> registers (no LDS, no waits forced before compute)
// ---------------------------------------------------------------------------
static __device__ __forceinline__ void stage_load(StageRegs& r,
    const void* __restrict__ Aptr, const float* __restrict__ W,
    int a_bf16, int tid, int mBase, int nBase, int M, int Nn, int K, int k0)
{
    const bool fullTile = (mBase + 128 <= M);
    if (!a_bf16) {
        const float* Af = (const float*)Aptr;
        #pragma unroll
        for (int i = 0; i < 4; ++i) {
            int f  = tid + i * 256;            // 0..1023 float4 slots (128 rows x 8)
            int rr = f >> 3;
            int c4 = f & 7;
            int gm = mBase + rr;
            float4 v = make_float4(0.f, 0.f, 0.f, 0.f);
            if (fullTile || gm < M) {
                const float* p = Af + (size_t)gm * K + k0 + c4 * 4;
                v = *(const float4*)p;
                if (k0 + 64 < K)
                    __builtin_prefetch(p + 64, 0, 0);   // global_prefetch_b8, next-next tile
            }
            r.a[i] = make_uint4(__float_as_uint(v.x), __float_as_uint(v.y),
                                __float_as_uint(v.z), __float_as_uint(v.w));
        }
    } else {
        const ushort_t* Ah = (const ushort_t*)Aptr;
        #pragma unroll
        for (int i = 0; i < 4; ++i) {
            int f  = tid + i * 256;            // each slot = 4 bf16
            int rr = f >> 3;
            int c4 = f & 7;
            int gm = mBase + rr;
            uint2 v = make_uint2(0u, 0u);
            if (fullTile || gm < M) {
                const ushort_t* p = Ah + (size_t)gm * K + k0 + c4 * 4;
                v = *(const uint2*)p;
                if (k0 + 64 < K)
                    __builtin_prefetch(p + 64, 0, 0);
            }
            r.a[i].x = v.x; r.a[i].y = v.y; r.a[i].z = 0u; r.a[i].w = 0u;
        }
    }
    #pragma unroll
    for (int i = 0; i < 2; ++i) {
        int f  = tid + i * 256;                // 0..511 float4 slots (32 rows x 16)
        int rr = f >> 4;
        int c4 = f & 15;
        r.b[i] = *(const float4*)(W + (size_t)(k0 + rr) * Nn + nBase + c4 * 4);
    }
}

// ---------------------------------------------------------------------------
// Stage phase 2: registers -> LDS (bf16, padded rows: 34 bf16 = 17 dwords)
// ---------------------------------------------------------------------------
static __device__ __forceinline__ void stage_store(uint_t* __restrict__ sA,
    uint_t* __restrict__ sB, const StageRegs& r, int a_bf16, int tid)
{
    ushort_t* sB16 = (ushort_t*)sB;
    if (!a_bf16) {
        #pragma unroll
        for (int i = 0; i < 4; ++i) {
            int f  = tid + i * 256;
            int rr = f >> 3;
            int c4 = f & 7;
            int d  = rr * 17 + c4 * 2;
            sA[d]     = (uint_t)bf_from_bits(r.a[i].x) | ((uint_t)bf_from_bits(r.a[i].y) << 16);
            sA[d + 1] = (uint_t)bf_from_bits(r.a[i].z) | ((uint_t)bf_from_bits(r.a[i].w) << 16);
        }
    } else {
        #pragma unroll
        for (int i = 0; i < 4; ++i) {
            int f  = tid + i * 256;
            int rr = f >> 3;
            int c4 = f & 7;
            int d  = rr * 17 + c4 * 2;
            sA[d]     = r.a[i].x;
            sA[d + 1] = r.a[i].y;
        }
    }
    #pragma unroll
    for (int i = 0; i < 2; ++i) {
        int f  = tid + i * 256;
        int rr = f >> 4;
        int c4 = f & 15;
        int n0 = c4 * 4;
        sB16[(n0 + 0) * 34 + rr] = f2bf(r.b[i].x);
        sB16[(n0 + 1) * 34 + rr] = f2bf(r.b[i].y);
        sB16[(n0 + 2) * 34 + rr] = f2bf(r.b[i].z);
        sB16[(n0 + 3) * 34 + rr] = f2bf(r.b[i].w);
    }
}

// ---------------------------------------------------------------------------
// GEMM: Out[M,N] = act(A[M,K] @ W[K,N] + bias) * scale
// Block 256 thr = 8 waves, tile 128x64, BK=32, double-buffered LDS.
// Each wave: 32x32 = 2x2 WMMA f32_16x16x32_bf16 tiles.
// ---------------------------------------------------------------------------
__global__ __launch_bounds__(256)
void gemm_bias_act(const void* __restrict__ Aptr,
                   const float* __restrict__ W,
                   const float* __restrict__ bias,
                   void* __restrict__ Outptr,
                   int M, int Nn, int K,
                   int a_bf16, int relu, float scale, int out_bf16)
{
    __shared__ uint_t sA0[128 * 17];
    __shared__ uint_t sB0[64 * 17];
    __shared__ uint_t sA1[128 * 17];
    __shared__ uint_t sB1[64 * 17];

    const int tid   = threadIdx.x;
    const int lane  = tid & 31;
    const int wav   = tid >> 5;
    const int wm    = wav >> 1;       // 0..3 (M)
    const int wn    = wav & 1;        // 0..1 (N)
    const int lhalf = lane >> 4;
    const int l15   = lane & 15;

    const int nBase = blockIdx.x * 64;
    const int mBase = blockIdx.y * 128;

    v8f acc[2][2];
    #pragma unroll
    for (int i = 0; i < 2; ++i)
        #pragma unroll
        for (int j = 0; j < 2; ++j)
            #pragma unroll
            for (int e = 0; e < 8; ++e)
                acc[i][j][e] = 0.0f;

    StageRegs regs;
    stage_load(regs, Aptr, W, a_bf16, tid, mBase, nBase, M, Nn, K, 0);

    for (int k0 = 0; k0 < K; k0 += 32) {
        const int cur = (k0 >> 5) & 1;
        uint_t* sA = cur ? sA1 : sA0;
        uint_t* sB = cur ? sB1 : sB0;

        // drain current tile's staged registers into LDS buffer `cur`
        stage_store(sA, sB, regs, a_bf16, tid);
        __syncthreads();

        // kick off global loads for the next tile (latency hidden by WMMAs)
        if (k0 + 32 < K)
            stage_load(regs, Aptr, W, a_bf16, tid, mBase, nBase, M, Nn, K, k0 + 32);

        // ---- fragments per ISA 7.12.2 layouts ----
        FragBF a0, a1, b0, b1;
        const uint_t* pa0 = sA + (wm * 32 + l15) * 17;
        const uint_t* pa1 = pa0 + 16 * 17;
        #pragma unroll
        for (int c = 0; c < 4; ++c) {
            a0.u[c]     = pa0[lhalf * 4 + c];
            a0.u[4 + c] = pa0[8 + lhalf * 4 + c];
            a1.u[c]     = pa1[lhalf * 4 + c];
            a1.u[4 + c] = pa1[8 + lhalf * 4 + c];
        }
        const uint_t* pb0 = sB + (wn * 32 + l15) * 17;
        const uint_t* pb1 = pb0 + 16 * 17;
        #pragma unroll
        for (int c = 0; c < 8; ++c) {
            b0.u[c] = pb0[(c >> 2) * 8 + lhalf * 4 + (c & 3)];
            b1.u[c] = pb1[(c >> 2) * 8 + lhalf * 4 + (c & 3)];
        }

        acc[0][0] = __builtin_amdgcn_wmma_f32_16x16x32_bf16(false, a0.v, false, b0.v, (short)0, acc[0][0], false, false);
        acc[0][1] = __builtin_amdgcn_wmma_f32_16x16x32_bf16(false, a0.v, false, b1.v, (short)0, acc[0][1], false, false);
        acc[1][0] = __builtin_amdgcn_wmma_f32_16x16x32_bf16(false, a1.v, false, b0.v, (short)0, acc[1][0], false, false);
        acc[1][1] = __builtin_amdgcn_wmma_f32_16x16x32_bf16(false, a1.v, false, b1.v, (short)0, acc[1][1], false, false);
    }

    // ---- epilogue: bias, act, scale, store (C layout: VGPR r -> M=r+8*lhalf) ----
    #pragma unroll
    for (int i = 0; i < 2; ++i) {
        #pragma unroll
        for (int j = 0; j < 2; ++j) {
            int col = nBase + wn * 32 + j * 16 + l15;
            float bia = bias[col];
            int rowBase = mBase + wm * 32 + i * 16 + lhalf * 8;
            #pragma unroll
            for (int r = 0; r < 8; ++r) {
                int row = rowBase + r;
                if (row < M) {
                    float vv = acc[i][j][r] + bia;
                    if (relu) vv = fmaxf(vv, 0.0f);
                    vv *= scale;
                    if (out_bf16)
                        ((ushort_t*)Outptr)[(size_t)row * Nn + col] = f2bf(vv);
                    else
                        ((float*)Outptr)[(size_t)row * Nn + col] = vv;
                }
            }
        }
    }
}

// ---------------------------------------------------------------------------
// logits[n,h] = dot(q[rb[n],h,:], k[rf[n],h,:])  (scaling already baked into q)
// ---------------------------------------------------------------------------
__global__ __launch_bounds__(256)
void logits_kernel(const float* __restrict__ q, const float* __restrict__ k,
                   const int* __restrict__ rb, const int* __restrict__ rf,
                   float* __restrict__ logits, int N)
{
    int t = blockIdx.x * 256 + threadIdx.x;
    if (t >= N * 8) return;
    int n = t >> 3, h = t & 7;
    const float4* qp = (const float4*)(q + (size_t)rb[n] * EMBED + h * HDIM);
    const float4* kp = (const float4*)(k + (size_t)rf[n] * EMBED + h * HDIM);
    float acc = 0.f;
    #pragma unroll
    for (int i = 0; i < 8; ++i) {
        float4 a = qp[i], b = kp[i];
        acc += a.x * b.x + a.y * b.y + a.z * b.z + a.w * b.w;
    }
    logits[t] = acc;
}

// ---------------------------------------------------------------------------
// Per-(key,head) softmax over its segment; W_sm rows padded with 0 past L.
// ---------------------------------------------------------------------------
__global__ __launch_bounds__(256)
void softmax_kernel(const float* __restrict__ logits,
                    const int* __restrict__ starts, const int* __restrict__ lens,
                    float* __restrict__ wsm, int ref_num)
{
    int t = blockIdx.x * 256 + threadIdx.x;
    if (t >= KLEN * HEADS) return;
    int f = t >> 3, h = t & 7;
    int s = starts[f], L = lens[f];
    float* orow = wsm + ((size_t)f * HEADS + h) * ref_num;
    if (L <= 0) {
        float inv = 1.0f / (float)ref_num;
        for (int j = 0; j < ref_num; ++j) orow[j] = inv;
        return;
    }
    float m = -INFINITY;
    for (int j = 0; j < L; ++j)
        m = fmaxf(m, logits[(size_t)(s + j) * HEADS + h]);
    float sum = 0.f;
    for (int j = 0; j < L; ++j)
        sum += expf(logits[(size_t)(s + j) * HEADS + h] - m);
    float inv = 1.0f / sum;
    for (int j = 0; j < L; ++j)
        orow[j] = expf(logits[(size_t)(s + j) * HEADS + h] - m) * inv;
    for (int j = L; j < ref_num; ++j) orow[j] = 0.f;
}

// ---------------------------------------------------------------------------
__global__ __launch_bounds__(256)
void zero_kernel(float4* __restrict__ out)
{
    size_t t = (size_t)blockIdx.x * 256 + threadIdx.x;
    out[t] = make_float4(0.f, 0.f, 0.f, 0.f);
}

// ---------------------------------------------------------------------------
// segment-sum scatter: out[rb[n], :] += w[n,h] * v[rf[n], :]   (64 thr / pair)
// ---------------------------------------------------------------------------
__global__ __launch_bounds__(256)
void scatter_kernel(const float* __restrict__ wsm, const float* __restrict__ value,
                    const int* __restrict__ rf, const int* __restrict__ rb,
                    const int* __restrict__ starts,
                    float* __restrict__ out, int N, int ref_num)
{
    size_t t = (size_t)blockIdx.x * 256 + threadIdx.x;
    if (t >= (size_t)N * 64) return;
    int n    = (int)(t >> 6);
    int part = (int)(t & 63);
    int f = rf[n];
    int b = rb[n];
    int j = n - starts[f];
    int h = part >> 3;
    float w = wsm[((size_t)f * HEADS + h) * ref_num + j];
    float4 vv = *(const float4*)(value + (size_t)f * EMBED + part * 4);
    float* op = out + (size_t)b * EMBED + part * 4;
    atomicAdd(op + 0, w * vv.x);
    atomicAdd(op + 1, w * vv.y);
    atomicAdd(op + 2, w * vv.z);
    atomicAdd(op + 3, w * vv.w);
}

// ---------------------------------------------------------------------------
// in-place LayerNorm, one wave32 per 256-float row
// ---------------------------------------------------------------------------
__global__ __launch_bounds__(256)
void ln_kernel(float* __restrict__ out, const float* __restrict__ g,
               const float* __restrict__ b)
{
    int wave = (blockIdx.x * 256 + threadIdx.x) >> 5;
    int lane = threadIdx.x & 31;
    if (wave >= TGT) return;
    float* row = out + (size_t)wave * EMBED;
    float4 a0 = *(float4*)(row + lane * 8);
    float4 a1 = *(float4*)(row + lane * 8 + 4);
    float s  = a0.x + a0.y + a0.z + a0.w + a1.x + a1.y + a1.z + a1.w;
    float ss = a0.x*a0.x + a0.y*a0.y + a0.z*a0.z + a0.w*a0.w
             + a1.x*a1.x + a1.y*a1.y + a1.z*a1.z + a1.w*a1.w;
    #pragma unroll
    for (int off = 16; off > 0; off >>= 1) {
        s  += __shfl_xor(s,  off, 32);
        ss += __shfl_xor(ss, off, 32);
    }
    float mu   = s * (1.0f / EMBED);
    float var  = ss * (1.0f / EMBED) - mu * mu;
    float rstd = rsqrtf(var + 1e-5f);
    int c = lane * 8;
    float4 g0 = *(const float4*)(g + c),     g1 = *(const float4*)(g + c + 4);
    float4 b0 = *(const float4*)(b + c),     b1 = *(const float4*)(b + c + 4);
    a0.x = (a0.x - mu) * rstd * g0.x + b0.x;
    a0.y = (a0.y - mu) * rstd * g0.y + b0.y;
    a0.z = (a0.z - mu) * rstd * g0.z + b0.z;
    a0.w = (a0.w - mu) * rstd * g0.w + b0.w;
    a1.x = (a1.x - mu) * rstd * g1.x + b1.x;
    a1.y = (a1.y - mu) * rstd * g1.y + b1.y;
    a1.z = (a1.z - mu) * rstd * g1.z + b1.z;
    a1.w = (a1.w - mu) * rstd * g1.w + b1.w;
    *(float4*)(row + lane * 8)     = a0;
    *(float4*)(row + lane * 8 + 4) = a1;
}

// ---------------------------------------------------------------------------
extern "C" void kernel_launch(void* const* d_in, const int* in_sizes, int n_in,
                              void* d_out, int out_size, void* d_ws, size_t ws_size,
                              hipStream_t stream)
{
    const float* query   = (const float*)d_in[0];
    const float* key_in  = (const float*)d_in[1];
    const float* value   = (const float*)d_in[2];
    const int*   rfeat   = (const int*)d_in[3];
    const int*   rbev    = (const int*)d_in[4];
    const int*   starts  = (const int*)d_in[5];
    const int*   lens    = (const int*)d_in[6];
    const float* wq1 = (const float*)d_in[7];
    const float* bq1 = (const float*)d_in[8];
    const float* wq2 = (const float*)d_in[9];
    const float* bq2 = (const float*)d_in[10];
    const float* wk1 = (const float*)d_in[11];
    const float* bk1 = (const float*)d_in[12];
    const float* wk2 = (const float*)d_in[13];
    const float* bk2 = (const float*)d_in[14];
    const float* ln_g = (const float*)d_in[15];
    const float* ln_b = (const float*)d_in[16];

    const int N = in_sizes[3];
    const int ref_num = (out_size - TGT * EMBED) / (KLEN * HEADS);

    // workspace layout (bytes)
    const size_t HQ_BYTES = (size_t)TGT  * 512 * 2;   // hq bf16 (later reused for k f32)
    const size_t Q_BYTES  = (size_t)TGT  * EMBED * 4; // q f32
    const size_t HK_BYTES = (size_t)KLEN * 512 * 2;   // hk bf16

    char* ws = (char*)d_ws;
    ushort_t* hq     = (ushort_t*)ws;
    float*    qf     = (float*)(ws + HQ_BYTES);
    ushort_t* hk     = (ushort_t*)(ws + HQ_BYTES + Q_BYTES);
    float*    kf     = (float*)ws;                               // reuse hq region
    float*    logits = (float*)(ws + HQ_BYTES + Q_BYTES + HK_BYTES);

    float* outAcc = (float*)d_out;                               // TGT*EMBED
    float* wsm    = (float*)d_out + (size_t)TGT * EMBED;         // KLEN*HEADS*ref_num

    const float scaling = 1.0f / sqrtf((float)HDIM);

    // 1) hq = relu(query @ wq1 + bq1)  [bf16]
    gemm_bias_act<<<dim3(512 / 64, (TGT + 127) / 128), 256, 0, stream>>>(
        query, wq1, bq1, hq, TGT, 512, 256, 0, 1, 1.0f, 1);
    // 2) q = (hq @ wq2 + bq2) * scaling  [f32]
    gemm_bias_act<<<dim3(EMBED / 64, (TGT + 127) / 128), 256, 0, stream>>>(
        hq, wq2, bq2, qf, TGT, EMBED, 512, 1, 0, scaling, 0);
    // 3) hk = relu(key_in @ wk1 + bk1)  [bf16]
    gemm_bias_act<<<dim3(512 / 64, (KLEN + 127) / 128), 256, 0, stream>>>(
        key_in, wk1, bk1, hk, KLEN, 512, 256, 0, 1, 1.0f, 1);
    // 4) k = hk @ wk2 + bk2  [f32]  (overwrites hq region; hq no longer needed)
    gemm_bias_act<<<dim3(EMBED / 64, (KLEN + 127) / 128), 256, 0, stream>>>(
        hk, wk2, bk2, kf, KLEN, EMBED, 512, 1, 0, 1.0f, 0);

    // 5) gathered logits
    logits_kernel<<<(N * 8 + 255) / 256, 256, 0, stream>>>(qf, kf, rbev, rfeat, logits, N);

    // 6) per-(key,head) segment softmax -> W_sm (output #2)
    softmax_kernel<<<(KLEN * HEADS + 255) / 256, 256, 0, stream>>>(logits, starts, lens, wsm, ref_num);

    // 7) zero accumulator (output #1 region), 8100 blocks exact
    zero_kernel<<<(TGT * EMBED / 4) / 256, 256, 0, stream>>>((float4*)outAcc);

    // 8) scatter segment-sum with float atomics
    {
        long long total = (long long)N * 64;
        int blocks = (int)((total + 255) / 256);
        scatter_kernel<<<blocks, 256, 0, stream>>>(wsm, value, rfeat, rbev, starts, outAcc, N, ref_num);
    }

    // 9) in-place LayerNorm (one wave per row; 8 waves per block)
    ln_kernel<<<(TGT + 7) / 8, 256, 0, stream>>>(outAcc, ln_g, ln_b);
}
